// MultiHeadAttention_75127567941993
// MI455X (gfx1250) — matrix operand order repaired and checked by reading, not compiled, not run on hardware
//
#include <hip/hip_runtime.h>
#include <hip/hip_bf16.h>

#define DEVINL __device__ __forceinline__

typedef __bf16 bf16_t;
typedef __attribute__((ext_vector_type(16))) __bf16 v16bf;
typedef __attribute__((ext_vector_type(8)))  float  v8f;

static constexpr int Bc = 2;        // batch
static constexpr int Sc = 2048;     // sequence
static constexpr int Ec = 1024;     // embed
static constexpr int Hc = 16;       // heads
static constexpr int Dc = 64;       // head dim
static constexpr int Mc = Bc * Sc;  // 4096 rows

union Frag16 { v16bf v; uint4 q[2]; };

// Load a 16-element bf16 fragment as two 16B chunks (per the CDNA5 A/B VGPR layouts).
DEVINL v16bf load_frag_split(const bf16_t* p0, const bf16_t* p1) {
    Frag16 f;
    f.q[0] = *(const uint4*)p0;
    f.q[1] = *(const uint4*)p1;
    return f.v;
}

DEVINL v8f wmma_bf16(v16bf a, v16bf b, v8f c) {
    // (neg_a, A, neg_b, B, c_mod, C, reuse_a, reuse_b)
    return __builtin_amdgcn_wmma_f32_16x16x32_bf16(false, a, false, b, (short)0, c,
                                                   false, false);
}

// CDNA5 async global->LDS copy (CDNA5 ISA): VDST = LDS byte
// address (low 32 bits of a generic __shared__ pointer), VADDR = 64-bit global
// address. Tracked with ASYNCcnt.
DEVINL void async_ld_b128(unsigned lds_addr, const void* gptr) {
    asm volatile("global_load_async_to_lds_b128 %0, %1, off"
                 :: "v"(lds_addr), "v"(gptr)
                 : "memory");
}

// DPP16 row-rotate (VALU cross-lane, no LDS). CTRL: 0x120 | n = row_ror:n.
template <int CTRL>
DEVINL float dpp_ror(float x) {
    int xi = __float_as_int(x);
    int r  = __builtin_amdgcn_update_dpp(xi, xi, CTRL, 0xf, 0xf, true);
    return __int_as_float(r);
}

// Max over the 16-lane row group via circular rotate-reduce (1,2,4,8).
DEVINL float rowmax16(float x) {
    x = fmaxf(x, dpp_ror<0x121>(x));
    x = fmaxf(x, dpp_ror<0x122>(x));
    x = fmaxf(x, dpp_ror<0x124>(x));
    x = fmaxf(x, dpp_ror<0x128>(x));
    return x;
}

// ---------------------------------------------------------------------------
// f32 -> bf16 conversion, 4 elements/thread
// ---------------------------------------------------------------------------
struct BF4 { bf16_t x, y, z, w; };

__global__ __launch_bounds__(256)
void cvt_f32_bf16_kernel(const float* __restrict__ src, bf16_t* __restrict__ dst, int n4) {
    int i = blockIdx.x * blockDim.x + threadIdx.x;
    if (i >= n4) return;
    float4 f = ((const float4*)src)[i];
    BF4 o { (bf16_t)f.x, (bf16_t)f.y, (bf16_t)f.z, (bf16_t)f.w };
    ((BF4*)dst)[i] = o;
}

// ---------------------------------------------------------------------------
// GEMM: out[m,n] = (sum_k A[m,k]*W[n,k] + bias[n]) * scale
// A: [M,K] bf16 row-major; W: [N,K] bf16 row-major (i.e. X @ W^T).
// Block = 256 threads = 8 waves (4 M-waves x 2 N-waves), wave tile 32x32,
// block tile 128(M) x 64(N). K stepped by 32 (one WMMA K-chunk). Weights are
// L2-resident (2 MB bf16 vs 192 MB L2), so direct 32B/lane global frag loads
// feed 8192 FLOP of WMMA each.
// LAYOUT: 0 = bf16 head layout [B,H,S,D]; 1 = bf16 transposed head [B,H,D,S];
//         2 = f32 row-major [M,N].
// ---------------------------------------------------------------------------
template <int LAYOUT>
__global__ __launch_bounds__(256)
void gemm_bf16_kernel(const bf16_t* __restrict__ A, const bf16_t* __restrict__ W,
                      const float* __restrict__ bias, void* __restrict__ out,
                      int K, int N, float scale) {
    const int lane = threadIdx.x & 31;
    const int wave = threadIdx.x >> 5;
    const int hs   = lane >> 4;   // lane-half select
    const int ln   = lane & 15;

    const int m0 = blockIdx.y * 128 + (wave & 3) * 32;
    const int n0 = blockIdx.x * 64  + (wave >> 2) * 32;

    const bf16_t* arow[2];
    const bf16_t* wrow[2];
#pragma unroll
    for (int t = 0; t < 2; ++t) {
        arow[t] = A + (size_t)(m0 + t * 16 + ln) * K;   // A frag: row = lane&15
        wrow[t] = W + (size_t)(n0 + t * 16 + ln) * K;   // B frag: col = lane&15
    }

    v8f acc[2][2] = {};
    for (int k = 0; k < K; k += 32) {
        v16bf af[2], bw[2];
#pragma unroll
        for (int t = 0; t < 2; ++t) {
            // A 16x32 bf16 frag: elems 0..7 = K(k+hs*8+i), 8..15 = K(k+16+hs*8+i)
            af[t] = load_frag_split(arow[t] + k + hs * 8,
                                    arow[t] + k + 16 + hs * 8);
            // B 32x16 bf16 frag: elems 0..7 = K(k+hs*8+i), 8..15 = K(k+16+hs*8+i)
            bw[t] = load_frag_split(wrow[t] + k + hs * 8,
                                    wrow[t] + k + 16 + hs * 8);
        }
#pragma unroll
        for (int tm = 0; tm < 2; ++tm)
#pragma unroll
            for (int tn = 0; tn < 2; ++tn)
                acc[tm][tn] = wmma_bf16(af[tm], bw[tn], acc[tm][tn]);
    }

    // Epilogue: C/D layout -> element (r,lane) is (M = r + 8*hs, N = lane&15)
#pragma unroll
    for (int tm = 0; tm < 2; ++tm)
#pragma unroll
        for (int tn = 0; tn < 2; ++tn)
#pragma unroll
            for (int r = 0; r < 8; ++r) {
                const int row = m0 + tm * 16 + r + 8 * hs;
                const int col = n0 + tn * 16 + ln;
                const float val = (acc[tm][tn][r] + bias[col]) * scale;
                if (LAYOUT == 0) {            // [B,H,S,D] bf16
                    const int b = row >> 11, s = row & (Sc - 1);
                    const int h = col >> 6,  d = col & (Dc - 1);
                    ((bf16_t*)out)[(((size_t)(b * Hc + h) * Sc + s) * Dc) + d] = (bf16_t)val;
                } else if (LAYOUT == 1) {     // [B,H,D,S] bf16 (V transposed)
                    const int b = row >> 11, s = row & (Sc - 1);
                    const int h = col >> 6,  d = col & (Dc - 1);
                    ((bf16_t*)out)[(((size_t)(b * Hc + h) * Dc + d) * Sc) + s] = (bf16_t)val;
                } else {                      // [M,N] f32
                    ((float*)out)[(size_t)row * N + col] = val;
                }
            }
}

// ---------------------------------------------------------------------------
// Flash attention core.
// Block = 8 waves, all sharing one (b,h); wave w owns the 16-row Q tile at
// q0 = blockIdx.x*128 + w*16. Key tiles of 32 are staged into LDS once per
// BLOCK with double-buffered global_load_async_to_lds_b128 (ASYNCcnt), so the
// next tile's HBM/L2 traffic overlaps the current tile's WMMAs and K/V global
// traffic drops 8x vs per-wave loads.
// Per key tile and wave:
//   4 WMMA for S = Q K^T, online softmax (row max via DPP row_ror reductions,
//   pure VALU that co-executes with XDL WMMA), P staged through per-wave LDS
//   to re-layout C->A fragment, 1 WMMA (P x ones) for the row sums,
//   4 WMMA for O += P V.  Output written bf16 into merged layout att[B,S,H*D].
// ---------------------------------------------------------------------------
__global__ __launch_bounds__(256)
void flash_attn_kernel(const bf16_t* __restrict__ Qh, const bf16_t* __restrict__ Kh,
                       const bf16_t* __restrict__ Vt, bf16_t* __restrict__ att) {
    const int tid  = threadIdx.x;
    const int lane = tid & 31;
    const int wave = tid >> 5;
    const int hs   = lane >> 4;
    const int ln   = lane & 15;

    const int b  = blockIdx.z;
    const int h  = blockIdx.y;
    const int q0 = (blockIdx.x * 8 + wave) * 16;
    const size_t bh = (size_t)(b * Hc + h);

    const bf16_t* Qb = Qh + bh * Sc * Dc;
    const bf16_t* Kb = Kh + bh * Sc * Dc;
    const bf16_t* Vb = Vt + bh * Dc * Sc;

    // LDS: double-buffered K (32x64) and V (64x32) tiles + per-wave P tile.
    __shared__ __align__(128) bf16_t ldsK[2][32 * 64];   // 2 x 4 KB
    __shared__ __align__(128) bf16_t ldsV[2][64 * 32];   // 2 x 4 KB
    __shared__ __align__(64)  bf16_t lds_p[8][16 * 32];  // 8 x 1 KB
    bf16_t* P = &lds_p[wave][0];

    // Per-thread async-copy addresses: thread t moves one 16B chunk of K and
    // one of V per tile. K tile is fully contiguous (32 rows x 128B = 4 KB).
    // V is [D,S], so row d of the tile is 64B at stride S*2.
    const unsigned ldsK_a[2] = { (unsigned)(uintptr_t)&ldsK[0][0] + (unsigned)tid * 16u,
                                 (unsigned)(uintptr_t)&ldsK[1][0] + (unsigned)tid * 16u };
    const int vd = tid >> 2, vc = tid & 3;
    const unsigned ldsV_a[2] = { (unsigned)(uintptr_t)&ldsV[0][0] + (unsigned)(vd * 64 + vc * 16),
                                 (unsigned)(uintptr_t)&ldsV[1][0] + (unsigned)(vd * 64 + vc * 16) };
    const char* gK0 = (const char*)Kb + (size_t)tid * 16u;           // + key0*128 per tile
    const char* gV0 = (const char*)(Vb + (size_t)vd * Sc) + vc * 16; // + key0*2 per tile

    // Q A-fragments for both K-chunks of D=64 (resident for the whole loop)
    v16bf aq[2];
#pragma unroll
    for (int kk = 0; kk < 2; ++kk) {
        const bf16_t* qp = Qb + (size_t)(q0 + ln) * Dc + kk * 32;
        aq[kk] = load_frag_split(qp + hs * 8, qp + 16 + hs * 8);
    }

    // all-ones B fragment: wmma(P, ones) -> row sums of P replicated per lane
    v16bf ones;
#pragma unroll
    for (int e = 0; e < 16; ++e) ones[e] = (bf16_t)1.0f;

    float mi[8], li[8];
    v8f acc[4] = {};
#pragma unroll
    for (int r = 0; r < 8; ++r) { mi[r] = -1e30f; li[r] = 0.0f; }

    constexpr int NT = Sc / 32;   // 64 key tiles

    // Prologue: stage tile 0 into buffer 0.
    async_ld_b128(ldsK_a[0], gK0);
    async_ld_b128(ldsV_a[0], gV0);

    for (int kt = 0; kt < NT; ++kt) {
        const int buf = kt & 1;
        // Prefetch next tile into the other buffer (overlaps with compute).
        if (kt + 1 < NT) {
            const size_t nk = (size_t)(kt + 1) * 32;
            async_ld_b128(ldsK_a[buf ^ 1], gK0 + nk * (Dc * 2));
            async_ld_b128(ldsV_a[buf ^ 1], gV0 + nk * 2);
            asm volatile("s_wait_asynccnt 0x2" ::: "memory");  // current tile done
        } else {
            asm volatile("s_wait_asynccnt 0x0" ::: "memory");
        }
        __syncthreads();   // every thread's chunk of the current tile is in LDS

        const bf16_t* Kt = &ldsK[buf][0];
        const bf16_t* Vtile = &ldsV[buf][0];

        // --- scores: S(16x32) = Q(16x64) * K^T(64x32), K from LDS ---
        v8f sc[2] = {};
#pragma unroll
        for (int tn = 0; tn < 2; ++tn) {
#pragma unroll
            for (int kk = 0; kk < 2; ++kk) {
                const bf16_t* kp = Kt + (tn * 16 + ln) * Dc + kk * 32 + hs * 8;
                v16bf bk = load_frag_split(kp, kp + 16);
                sc[tn] = wmma_bf16(aq[kk], bk, sc[tn]);
            }
        }

        // --- online softmax; row m = r + 8*hs lives across the 16-lane group ---
        float corr[8];
#pragma unroll
        for (int r = 0; r < 8; ++r) {
            const float rm = rowmax16(fmaxf(sc[0][r], sc[1][r]));
            const float mnew = fmaxf(mi[r], rm);
            corr[r] = __expf(mi[r] - mnew);
            const float p0 = __expf(sc[0][r] - mnew);
            const float p1 = __expf(sc[1][r] - mnew);
            mi[r] = mnew;
#pragma unroll
            for (int t = 0; t < 4; ++t) acc[t][r] *= corr[r];
            const int prow = (r + 8 * hs) * 32;
            P[prow + ln]      = (bf16_t)p0;
            P[prow + 16 + ln] = (bf16_t)p1;
        }

        // wave-synchronous LDS handoff (cross-lane within the wave)
        asm volatile("s_wait_dscnt 0x0" ::: "memory");
        __builtin_amdgcn_wave_barrier();

        // re-load P in A-fragment layout
        const bf16_t* pp = P + ln * 32;
        v16bf ap = load_frag_split(pp + hs * 8, pp + 16 + hs * 8);

        // row sums of P via one WMMA against the all-ones matrix
        v8f zero = {};
        v8f lsum = wmma_bf16(ap, ones, zero);
#pragma unroll
        for (int r = 0; r < 8; ++r) li[r] = li[r] * corr[r] + lsum[r];

        // --- O(16x64) += P(16x32) * V(32x64); V tile from LDS, [D][keys] ---
#pragma unroll
        for (int t = 0; t < 4; ++t) {
            const bf16_t* vp = Vtile + (t * 16 + ln) * 32 + hs * 8;
            v16bf bv = load_frag_split(vp, vp + 16);
            acc[t] = wmma_bf16(ap, bv, acc[t]);
        }

        __syncthreads();   // all waves done reading before buffer is re-staged
    }

    // normalize and write merged head layout att[b, s, h*64 + d] (bf16)
#pragma unroll
    for (int t = 0; t < 4; ++t)
#pragma unroll
        for (int r = 0; r < 8; ++r) {
            const int srow = q0 + r + 8 * hs;
            const int col  = h * Dc + t * 16 + ln;
            att[((size_t)b * Sc + srow) * Ec + col] = (bf16_t)(acc[t][r] / li[r]);
        }
}

// ---------------------------------------------------------------------------
// Launch
// ---------------------------------------------------------------------------
extern "C" void kernel_launch(void* const* d_in, const int* in_sizes, int n_in,
                              void* d_out, int out_size, void* d_ws, size_t ws_size,
                              hipStream_t stream) {
    const float* q  = (const float*)d_in[0];
    const float* k  = (const float*)d_in[1];
    const float* v  = (const float*)d_in[2];
    const float* Wq = (const float*)d_in[3];
    const float* bq = (const float*)d_in[4];
    const float* Wk = (const float*)d_in[5];
    const float* bk = (const float*)d_in[6];
    const float* Wv = (const float*)d_in[7];
    const float* bv = (const float*)d_in[8];
    const float* Wo = (const float*)d_in[9];
    const float* bo = (const float*)d_in[10];

    char* ws = (char*)d_ws;
    const size_t MB = 1u << 20;
    bf16_t* qbf = (bf16_t*)(ws + 0 * MB);    // 8 MB each
    bf16_t* kbf = (bf16_t*)(ws + 8 * MB);
    bf16_t* vbf = (bf16_t*)(ws + 16 * MB);
    bf16_t* wqb = (bf16_t*)(ws + 24 * MB);   // 2 MB each
    bf16_t* wkb = (bf16_t*)(ws + 26 * MB);
    bf16_t* wvb = (bf16_t*)(ws + 28 * MB);
    bf16_t* wob = (bf16_t*)(ws + 30 * MB);
    bf16_t* Qh  = (bf16_t*)(ws + 32 * MB);   // [B,H,S,D] bf16
    bf16_t* Kh  = (bf16_t*)(ws + 40 * MB);   // [B,H,S,D] bf16
    bf16_t* Vt  = (bf16_t*)(ws + 48 * MB);   // [B,H,D,S] bf16
    bf16_t* att = (bf16_t*)(ws + 56 * MB);   // [B,S,E]   bf16

    const int nx = Mc * Ec;      // 4M activations
    const int nw = Ec * Ec;      // 1M weights
    cvt_f32_bf16_kernel<<<nx / 4 / 256, 256, 0, stream>>>(q,  qbf, nx / 4);
    cvt_f32_bf16_kernel<<<nx / 4 / 256, 256, 0, stream>>>(k,  kbf, nx / 4);
    cvt_f32_bf16_kernel<<<nx / 4 / 256, 256, 0, stream>>>(v,  vbf, nx / 4);
    cvt_f32_bf16_kernel<<<nw / 4 / 256, 256, 0, stream>>>(Wq, wqb, nw / 4);
    cvt_f32_bf16_kernel<<<nw / 4 / 256, 256, 0, stream>>>(Wk, wkb, nw / 4);
    cvt_f32_bf16_kernel<<<nw / 4 / 256, 256, 0, stream>>>(Wv, wvb, nw / 4);
    cvt_f32_bf16_kernel<<<nw / 4 / 256, 256, 0, stream>>>(Wo, wob, nw / 4);

    const dim3 gg(Ec / 64, Mc / 128);   // (16, 32)
    const float inv_sqrt_d = 0.125f;    // 1/sqrt(64), folded into Q projection
    gemm_bf16_kernel<0><<<gg, 256, 0, stream>>>(qbf, wqb, bq, (void*)Qh, Ec, Ec, inv_sqrt_d);
    gemm_bf16_kernel<0><<<gg, 256, 0, stream>>>(kbf, wkb, bk, (void*)Kh, Ec, Ec, 1.0f);
    gemm_bf16_kernel<1><<<gg, 256, 0, stream>>>(vbf, wvb, bv, (void*)Vt, Ec, Ec, 1.0f);

    flash_attn_kernel<<<dim3(Sc / 128, Hc, Bc), 256, 0, stream>>>(Qh, Kh, Vt, att);

    gemm_bf16_kernel<2><<<gg, 256, 0, stream>>>(att, wob, bo, d_out, Ec, Ec, 1.0f);
}
